// EncoderBlock_22402549416485
// MI455X (gfx1250) — matrix operand order repaired and checked by reading, not compiled, not run on hardware
//
#include <hip/hip_runtime.h>

#ifndef __has_builtin
#define __has_builtin(x) 0
#endif

// ---------------------------------------------------------------------------
// Types for CDNA5 WMMA (wave32): D(f32 16x16) = A(bf16 16x32) x B(bf16 32x16) + C
// ---------------------------------------------------------------------------
typedef __bf16 bf16;
typedef __attribute__((ext_vector_type(16))) __bf16 v16bf;
typedef __attribute__((ext_vector_type(8)))  __bf16 v8bf;
typedef __attribute__((ext_vector_type(8)))  float  v8f;

#define DEV static __device__ __forceinline__

DEV v8f zero8() {
  v8f z;
#pragma unroll
  for (int i = 0; i < 8; ++i) z[i] = 0.0f;
  return z;
}

DEV v8f wmma_bf16(v16bf a, v16bf b, v8f c) {
  // 8 args: (neg_a, A, neg_b, B, c_mod, C, reuse_a, reuse_b)
  return __builtin_amdgcn_wmma_f32_16x16x32_bf16(false, a, false, b, (short)0, c,
                                                 false, false);
}

// A-matrix fragment (16x32, row-major source, stride ld elements).
// ISA 7.12.2 16-bit A layout: lane L holds row M=L&15; lane half selects
// K offset {0|8}; VGPRs 0-3 cover K base+0..7, VGPRs 4-7 cover K base+16..23.
DEV v16bf load_frag_a(const bf16* base, int row0, int k0, int ld) {
  const int lane = threadIdx.x & 31;
  const bf16* p = base + (size_t)(row0 + (lane & 15)) * ld + k0 + ((lane >> 4) << 3);
  v8bf lo = *(const v8bf*)(p);
  v8bf hi = *(const v8bf*)(p + 16);
  v16bf r;
#pragma unroll
  for (int i = 0; i < 8; ++i) { r[i] = lo[i]; r[i + 8] = hi[i]; }
  return r;
}

// B-matrix fragment (32x16) loaded from B^T stored row-major [N][K]:
// lane L holds column N=L&15; lane half selects K offset {0|8}; elems 0-7
// cover K base+0..7, elems 8-15 cover K base+16..23 (same as A).
DEV v16bf load_frag_b(const bf16* baseT, int col0, int k0, int ld) {
  const int lane = threadIdx.x & 31;
  const bf16* p = baseT + (size_t)(col0 + (lane & 15)) * ld + k0 + ((lane >> 4) << 3);
  v8bf lo = *(const v8bf*)(p);
  v8bf hi = *(const v8bf*)(p + 16);
  v16bf r;
#pragma unroll
  for (int i = 0; i < 8; ++i) { r[i] = lo[i]; r[i + 8] = hi[i]; }
  return r;
}

// ---------------------------------------------------------------------------
// Tensor Data Mover (gfx1250): 2D tile load Global->LDS with LDS row padding.
// This toolchain exposes the 6-arg builtin:
//   (uint32x4 g0, int32x8 g1, int32x4 g2, int32x4 g3, int32x8 g4, i32 cpol)
// ---------------------------------------------------------------------------
#if __has_builtin(__builtin_amdgcn_tensor_load_to_lds)
#define HAVE_TDM 1
typedef unsigned int v4u __attribute__((ext_vector_type(4)));
typedef int v8i __attribute__((ext_vector_type(8)));
typedef int v4i __attribute__((ext_vector_type(4)));

// Load a tile_d1 x tile_d0 (rows x elems) bf16 tile starting at gaddr
// (row stride = stride0 elements) into LDS at lds_off, padding +4 DWORDs
// after every 32 DWORDs (=> 64 bf16 row lands on a 72-element padded row).
DEV void tdm_load_2d_bf16(const void* gaddr, unsigned lds_off,
                          unsigned tile_d0, unsigned tile_d1,
                          unsigned tensor_d0, unsigned tensor_d1,
                          unsigned long long stride0) {
  unsigned long long ga = (unsigned long long)gaddr;
  v4u g0;
  g0[0] = 1u;                                         // count=1 (user descriptor)
  g0[1] = lds_off;                                    // lds_addr (bytes)
  g0[2] = (unsigned)(ga & 0xffffffffull);             // global_addr[31:0]
  g0[3] = (unsigned)((ga >> 32) & 0x1ffffffull)       // global_addr[56:32]
          | (2u << 30);                               // type=2 ("image")
  v8i g1;
  unsigned d0 = (1u << 16)                            // data_size: 1 => 2 bytes
              | (1u << 20)                            // pad_enable
              | (4u << 22)                            // pad_interval: 4 => 32 DWORDs
              | (3u << 25);                           // pad_amount:   3 => 4 DWORDs
  g1[0] = (int)d0;
  g1[1] = (int)((tensor_d0 & 0xffffu) << 16);         // tensor_dim0[15:0]
  g1[2] = (int)(((tensor_d0 >> 16) & 0xffffu) | ((tensor_d1 & 0xffffu) << 16));
  g1[3] = (int)(((tensor_d1 >> 16) & 0xffffu) | ((tile_d0 & 0xffffu) << 16));
  g1[4] = (int)(tile_d1 & 0xffffu);                   // tile_dim1 (tile_dim2=0)
  g1[5] = (int)(unsigned)(stride0 & 0xffffffffull);   // tensor_dim0_stride[31:0]
  g1[6] = (int)(unsigned)((stride0 >> 32) & 0xffffull);
  g1[7] = 0;
  v4i g2 = {0, 0, 0, 0};
  v4i g3 = {0, 0, 0, 0};
  v8i g4 = {0, 0, 0, 0, 0, 0, 0, 0};
  __builtin_amdgcn_tensor_load_to_lds(g0, g1, g2, g3, g4, 0);
}
#else
#define HAVE_TDM 0
#endif

// ---------------------------------------------------------------------------
// Weight convert + transpose: W[K][N] f32 -> Wt[N][K] bf16
// ---------------------------------------------------------------------------
__global__ void transpose_bf16_kernel(const float* __restrict__ W,
                                      bf16* __restrict__ Wt, int K, int N) {
  size_t idx = (size_t)blockIdx.x * 256 + threadIdx.x;
  if (idx >= (size_t)K * N) return;
  int k = (int)(idx % (size_t)K);
  int n = (int)(idx / (size_t)K);
  Wt[idx] = (bf16)W[(size_t)k * N + n];
}

// ---------------------------------------------------------------------------
// LayerNorm (fp32 in) -> bf16 out.  One 256-thread block per row, D=1024.
// ---------------------------------------------------------------------------
__global__ __launch_bounds__(256, 1)
void ln_bf16_kernel(const float* __restrict__ x, const float* __restrict__ g,
                    const float* __restrict__ b, bf16* __restrict__ z) {
  const int row = blockIdx.x;
  const float* xr = x + (size_t)row * 1024;
  const int tid = threadIdx.x;
  float v[4], s = 0.f, ss = 0.f;
#pragma unroll
  for (int i = 0; i < 4; ++i) {
    v[i] = xr[tid + i * 256];
    s += v[i];
    ss += v[i] * v[i];
  }
#pragma unroll
  for (int off = 16; off >= 1; off >>= 1) {
    s  += __shfl_xor(s,  off, 32);
    ss += __shfl_xor(ss, off, 32);
  }
  __shared__ float sm[8], sm2[8];
  if ((tid & 31) == 0) { sm[tid >> 5] = s; sm2[tid >> 5] = ss; }
  __syncthreads();
  s = 0.f; ss = 0.f;
#pragma unroll
  for (int w = 0; w < 8; ++w) { s += sm[w]; ss += sm2[w]; }
  const float mu   = s * (1.0f / 1024.0f);
  const float var  = ss * (1.0f / 1024.0f) - mu * mu;
  const float rstd = rsqrtf(var + 1e-5f);
#pragma unroll
  for (int i = 0; i < 4; ++i) {
    int c = tid + i * 256;
    z[(size_t)row * 1024 + c] = (bf16)((v[i] - mu) * rstd * g[c] + b[c]);
  }
}

// ---------------------------------------------------------------------------
// Generic bf16 GEMM: C[M][N] = A[M][K] x Bt[N][K]^T  (+bias, +GELU, +residual)
// Block tile 128x256, 8 waves of 64x64 (4x4 accumulators), K-step 32.
// Epilogue specialized at compile time -> branch-free.
// ---------------------------------------------------------------------------
template <bool HAS_BIAS, bool GELU_ACT, bool HAS_RES, bool OUT_F32>
__global__ __launch_bounds__(256, 1)
void gemm_bf16_kernel(const bf16* __restrict__ A, const bf16* __restrict__ Bt,
                      int M, int N, int K,
                      const float* __restrict__ bias,
                      const float* resid,   // may alias Cf (in-place residual)
                      float* Cf, bf16* __restrict__ Cb) {
  const int wid  = threadIdx.x >> 5;
  const int lane = threadIdx.x & 31;
  const int wm = blockIdx.y * 128 + (wid >> 2) * 64;
  const int wn = blockIdx.x * 256 + (wid & 3) * 64;

  v8f acc[4][4];
#pragma unroll
  for (int mt = 0; mt < 4; ++mt)
#pragma unroll
    for (int nt = 0; nt < 4; ++nt) acc[mt][nt] = zero8();

  for (int k0 = 0; k0 < K; k0 += 32) {
    // speculative prefetch of the next K-slice (drop-on-fault, no bounds branch)
    __builtin_prefetch(A  + (size_t)(wm + (lane & 15)) * K + k0 + 64, 0, 1);
    __builtin_prefetch(Bt + (size_t)(wn + (lane & 15)) * K + k0 + 64, 0, 1);
    v16bf a[4], b[4];
#pragma unroll
    for (int mt = 0; mt < 4; ++mt) a[mt] = load_frag_a(A, wm + mt * 16, k0, K);
#pragma unroll
    for (int nt = 0; nt < 4; ++nt) b[nt] = load_frag_b(Bt, wn + nt * 16, k0, K);
#pragma unroll
    for (int mt = 0; mt < 4; ++mt)
#pragma unroll
      for (int nt = 0; nt < 4; ++nt)
        acc[mt][nt] = wmma_bf16(a[mt], b[nt], acc[mt][nt]);
  }

  // C layout: vgpr r -> row r + 8*(lane>>4); col = lane&15
  const int col16 = lane & 15;
  const int rbase = (lane >> 4) << 3;
#pragma unroll
  for (int mt = 0; mt < 4; ++mt) {
#pragma unroll
    for (int nt = 0; nt < 4; ++nt) {
#pragma unroll
      for (int r = 0; r < 8; ++r) {
        int row  = wm + mt * 16 + rbase + r;
        int coln = wn + nt * 16 + col16;
        float v = acc[mt][nt][r];
        if constexpr (HAS_BIAS) v += bias[coln];
        if constexpr (GELU_ACT) v = 0.5f * v * (1.0f + erff(v * 0.70710678118654752f));
        if constexpr (HAS_RES)  v += resid[(size_t)row * N + coln];
        if constexpr (OUT_F32)  Cf[(size_t)row * N + coln] = v;
        else                    Cb[(size_t)row * N + coln] = (bf16)v;
      }
    }
  }
}

// ---------------------------------------------------------------------------
// Flash attention. qkv bf16 [N*T][3072] (Q|K|V interleaved per row),
// mask f32 [N][T] additive over keys, ctx bf16 [N*T][1024].
// grid.x = n*16+h, grid.y = q-block of 128 rows; 8 waves x 16 q-rows each.
// K chunk staged via Tensor Data Mover (async-tensor path) when available.
// ---------------------------------------------------------------------------
__global__ __launch_bounds__(256, 1)
void flash_attn_kernel(const bf16* __restrict__ qkv, const float* __restrict__ mask,
                       bf16* __restrict__ ctx) {
  constexpr int T = 2048, LDQ = 3072;
  const int nh  = blockIdx.x;
  const int n   = nh >> 4, h = nh & 15;
  const int tid = threadIdx.x, wid = tid >> 5, lane = tid & 31;
  const int col16 = lane & 15;
  const int rbase = (lane >> 4) << 3;
  const int q0 = blockIdx.y * 128 + wid * 16;

  __shared__ __align__(16) bf16 Ks[64][72];       // K chunk, row-major [s][d]
  __shared__ __align__(16) bf16 Vt[64][72];       // V chunk, transposed [d][s]
  __shared__ __align__(16) bf16 Ps[8][16][72];    // per-wave P tile [q][s]

  const bf16* Qb = qkv + (size_t)n * T * LDQ + h * 64;
  const bf16* Kb = Qb + 1024;
  const bf16* Vb = Qb + 2048;
  const float* mrow = mask + (size_t)n * T;
#if HAVE_TDM
  const unsigned ks_lds_off = (unsigned)(uintptr_t)(void*)&Ks[0][0];
#endif

  v16bf aq[2];
#pragma unroll
  for (int kk = 0; kk < 2; ++kk) aq[kk] = load_frag_a(Qb, q0, kk * 32, LDQ);

  v8f o[4];
#pragma unroll
  for (int i = 0; i < 4; ++i) o[i] = zero8();
  float mi[8], li[8];
#pragma unroll
  for (int r = 0; r < 8; ++r) { mi[r] = -1e30f; li[r] = 0.f; }

  for (int s0 = 0; s0 < T; s0 += 64) {
    // ---- stage K chunk: TDM async tensor load (64x64 bf16, padded rows) ----
#if HAVE_TDM
    if (wid == 0) {
      tdm_load_2d_bf16(Kb + (size_t)s0 * LDQ, ks_lds_off,
                       /*tile_d0=*/64, /*tile_d1=*/64,
                       /*tensor_d0=*/3072, /*tensor_d1=*/(unsigned)T,
                       /*stride0=*/(unsigned long long)LDQ);
      __builtin_amdgcn_s_wait_tensorcnt(0);
    }
#else
    for (int c = tid; c < 512; c += 256) {
      int s = c >> 3, p4 = c & 7;
      *(uint4*)(&Ks[s][p4 * 8]) = *(const uint4*)(Kb + (size_t)(s0 + s) * LDQ + p4 * 8);
    }
#endif
    // ---- stage V chunk transposed (coalesced u16 reads along d) ----
    for (int e = tid; e < 4096; e += 256) {
      int s = e >> 6, d = e & 63;
      Vt[d][s] = Vb[(size_t)(s0 + s) * LDQ + d];
    }
    __syncthreads();

    // ---- S = (Q K^T) * dk^-0.5 + mask ----
    v8f sacc[4];
#pragma unroll
    for (int st = 0; st < 4; ++st) {
      v8f a = zero8();
#pragma unroll
      for (int kk = 0; kk < 2; ++kk) {
        v16bf bk = load_frag_b(&Ks[0][0], st * 16, kk * 32, 72);
        a = wmma_bf16(aq[kk], bk, a);
      }
      float mk = mrow[s0 + st * 16 + col16];
#pragma unroll
      for (int r = 0; r < 8; ++r) a[r] = a[r] * 0.125f + mk;
      sacc[st] = a;
    }

    // ---- online softmax: row max over chunk (16-lane shuffle reduce) ----
    float mnew[8];
#pragma unroll
    for (int r = 0; r < 8; ++r) {
      float v = fmaxf(fmaxf(sacc[0][r], sacc[1][r]), fmaxf(sacc[2][r], sacc[3][r]));
#pragma unroll
      for (int off = 1; off < 16; off <<= 1) v = fmaxf(v, __shfl_xor(v, off, 32));
      mnew[r] = fmaxf(mi[r], v);
    }

    float rs[8];
#pragma unroll
    for (int r = 0; r < 8; ++r) rs[r] = 0.f;
#pragma unroll
    for (int st = 0; st < 4; ++st) {
#pragma unroll
      for (int r = 0; r < 8; ++r) {
        float p = __expf(sacc[st][r] - mnew[r]);
        rs[r] += p;
        Ps[wid][rbase + r][st * 16 + col16] = (bf16)p;  // C-layout -> A-layout via LDS
      }
    }
#pragma unroll
    for (int r = 0; r < 8; ++r) {
#pragma unroll
      for (int off = 1; off < 16; off <<= 1) rs[r] += __shfl_xor(rs[r], off, 32);
      float alpha = __expf(mi[r] - mnew[r]);
      li[r] = li[r] * alpha + rs[r];
      mi[r] = mnew[r];
#pragma unroll
      for (int nt = 0; nt < 4; ++nt) o[nt][r] *= alpha;
    }

    // ---- O += P V  (P from wave-private LDS; same-wave DS ops are in-order) ----
    v16bf ap[2];
#pragma unroll
    for (int kk = 0; kk < 2; ++kk)
      ap[kk] = load_frag_a(&Ps[wid][0][0], 0, kk * 32, 72);
#pragma unroll
    for (int nt = 0; nt < 4; ++nt) {
#pragma unroll
      for (int kk = 0; kk < 2; ++kk) {
        v16bf bv = load_frag_b(&Vt[0][0], nt * 16, kk * 32, 72);
        o[nt] = wmma_bf16(ap[kk], bv, o[nt]);
      }
    }
    __syncthreads();
  }

  // ---- finalize: O / l, write ctx[N,T,H,DK] ----
#pragma unroll
  for (int r = 0; r < 8; ++r) {
    float inv = 1.0f / li[r];
    int q = q0 + rbase + r;
#pragma unroll
    for (int nt = 0; nt < 4; ++nt)
      ctx[(size_t)(n * T + q) * 1024 + h * 64 + nt * 16 + col16] =
          (bf16)(o[nt][r] * inv);
  }
}

// ---------------------------------------------------------------------------
// Host-side orchestration (all on `stream`, no allocations — graph-capturable)
// ---------------------------------------------------------------------------
extern "C" void kernel_launch(void* const* d_in, const int* in_sizes, int n_in,
                              void* d_out, int out_size, void* d_ws, size_t ws_size,
                              hipStream_t stream) {
  (void)in_sizes; (void)n_in; (void)out_size; (void)ws_size;
  const float* x    = (const float*)d_in[0];
  const float* mask = (const float*)d_in[1];
  const float* Wqkv = (const float*)d_in[2];
  const float* Wout = (const float*)d_in[3];
  const float* ln1g = (const float*)d_in[4];
  const float* ln1b = (const float*)d_in[5];
  const float* ln2g = (const float*)d_in[6];
  const float* ln2b = (const float*)d_in[7];
  const float* W1   = (const float*)d_in[8];
  const float* b1   = (const float*)d_in[9];
  const float* W2   = (const float*)d_in[10];
  const float* b2   = (const float*)d_in[11];
  float* out = (float*)d_out;

  constexpr int D = 1024, M = 8192;  // M = N*T = 4*2048
  char* w = (char*)d_ws;
  bf16* z     = (bf16*)w; w += (size_t)M * D * 2;        // 16.8 MB (reused for z2)
  bf16* qkv   = (bf16*)w; w += (size_t)M * 3 * D * 2;    // 50.3 MB
  bf16* ctx   = (bf16*)w; w += (size_t)M * D * 2;        // 16.8 MB
  bf16* hbuf  = (bf16*)w; w += (size_t)M * 4 * D * 2;    // 67.1 MB
  bf16* WqkvT = (bf16*)w; w += (size_t)3 * D * D * 2;    //  6.3 MB
  bf16* WoutT = (bf16*)w; w += (size_t)D * D * 2;        //  2.1 MB
  bf16* W1T   = (bf16*)w; w += (size_t)4 * D * D * 2;    //  8.4 MB
  bf16* W2T   = (bf16*)w; w += (size_t)4 * D * D * 2;    //  8.4 MB  (~176 MB total)

  // 1) weights -> bf16, transposed to [N][K] for contiguous B-fragments
  transpose_bf16_kernel<<<dim3((3u * D * D + 255) / 256), 256, 0, stream>>>(Wqkv, WqkvT, D, 3 * D);
  transpose_bf16_kernel<<<dim3((D * D + 255) / 256), 256, 0, stream>>>(Wout, WoutT, D, D);
  transpose_bf16_kernel<<<dim3((4u * D * D + 255) / 256), 256, 0, stream>>>(W1, W1T, D, 4 * D);
  transpose_bf16_kernel<<<dim3((4u * D * D + 255) / 256), 256, 0, stream>>>(W2, W2T, 4 * D, D);

  // 2) z1 = LN1(x)
  ln_bf16_kernel<<<M, 256, 0, stream>>>(x, ln1g, ln1b, z);
  // 3) qkv = z1 @ Wqkv                                   (out: bf16)
  gemm_bf16_kernel<false, false, false, false>
      <<<dim3(3 * D / 256, M / 128), 256, 0, stream>>>(
          z, WqkvT, M, 3 * D, D, nullptr, nullptr, nullptr, qkv);
  // 4) ctx = flash attention
  flash_attn_kernel<<<dim3(64, 16), 256, 0, stream>>>(qkv, mask, ctx);
  // 5) x1 = x + ctx @ Wout   -> d_out                    (out: f32, +resid)
  gemm_bf16_kernel<false, false, true, true>
      <<<dim3(D / 256, M / 128), 256, 0, stream>>>(
          ctx, WoutT, M, D, D, nullptr, x, out, nullptr);
  // 6) z2 = LN2(x1)
  ln_bf16_kernel<<<M, 256, 0, stream>>>(out, ln2g, ln2b, z);
  // 7) h = GELU(z2 @ W1 + b1)                            (out: bf16, +bias+GELU)
  gemm_bf16_kernel<true, true, false, false>
      <<<dim3(4 * D / 256, M / 128), 256, 0, stream>>>(
          z, W1T, M, 4 * D, D, b1, nullptr, nullptr, hbuf);
  // 8) out = x1 + (h @ W2 + b2)   (in-place residual into d_out)
  gemm_bf16_kernel<true, false, true, true>
      <<<dim3(D / 256, M / 128), 256, 0, stream>>>(
          hbuf, W2T, M, D, 4 * D, b2, out, out, nullptr);
}
